// GCNLayer_59219009077973
// MI455X (gfx1250) — compile-verified, matrix operand order checked by reading, not run on hardware
//
#include <hip/hip_runtime.h>
#include <hip/hip_bf16.h>

// ---------------------------------------------------------------------------
// GCN layer for MI455X (gfx1250):
//   h   = x @ W + bias                       (bf16 WMMA, f32 accumulate)
//   out = segment_sum(val * h[src] -> dst)   (L2-resident gather + f32 atomics)
//
// Roofline: GEMM = 13.1 GFLOP; HBM floor (x in + h out) ~205 MB -> ~8.8 us at
// 23.3 TB/s; only the 16-bit matrix pipe exceeds the required ~1.5 PFLOPS, so
// the GEMM runs on v_wmma_f32_16x16x32_bf16. B (W fragments, 128 KB) is staged
// through LDS (320 KB/WGP) in two 64 KB halves so all 8 waves of a block share
// it -> ds_load_b128 feeds the WMMAs instead of per-wave L2 traffic.
// SpMM gathers hit L2 (h = 102.4 MB < 192 MB) and scatter with
// global_atomic_add_f32.
// ---------------------------------------------------------------------------

typedef __attribute__((ext_vector_type(16))) __bf16 v16bf;
typedef __attribute__((ext_vector_type(8)))  float  v8f;
typedef __attribute__((ext_vector_type(4)))  float  v4f;

#define DDIM   256
#define KSTEPS 8      // 256 / 32
#define NTILES 16     // 256 / 16
#define NFRAG_HALFS (KSTEPS * NTILES * 32 * 16)   // 65536 bf16 = 128 KB

__device__ __forceinline__ __bf16 f2bf(float f) { return (__bf16)f; }

// ---------------------------------------------------------------------------
// Pre-swizzle W[256,256] (row-major f32) into per-lane WMMA B-fragments (bf16)
// in global scratch. Fragment layout per (kstep, ntile): lane l holds 16
// halfs; K-map (16-bit B 32x16, ISA 7.12.2): base = (l<16?0:8),
// elem e -> K = base + (e<8 ? e : e+8); N = ntile*16 + (l & 15).
// ---------------------------------------------------------------------------
__global__ void wfrag_prep_kernel(const float* __restrict__ W,
                                  __bf16* __restrict__ bfrag) {
    int idx = blockIdx.x * blockDim.x + threadIdx.x;
    if (idx >= NFRAG_HALFS) return;
    int e    = idx & 15;
    int lane = (idx >> 4) & 31;
    int nt   = (idx >> 9) & 15;
    int ks   = idx >> 13;
    int base = (lane < 16) ? 0 : 8;
    int k    = ks * 32 + base + ((e < 8) ? e : (e + 8));
    int n    = nt * 16 + (lane & 15);
    bfrag[idx] = f2bf(W[k * DDIM + n]);
}

// ---------------------------------------------------------------------------
// Zero d_out (harness poisons it; we accumulate with atomics, so each launch
// must start from zero -> deterministic across graph replays).
// ---------------------------------------------------------------------------
__global__ void zero_out_kernel(float4* __restrict__ out, long n4) {
    long i = (long)blockIdx.x * blockDim.x + threadIdx.x;
    if (i < n4) out[i] = make_float4(0.f, 0.f, 0.f, 0.f);
}

// ---------------------------------------------------------------------------
// GEMM: block = 8 waves; each wave computes a 16-row tile across all 16
// column tiles (16 v8f accumulators = 128 VGPRs). B fragments are staged
// block-cooperatively into 64 KB LDS, half of K at a time; per k-step one
// A-fragment load (4 x b128 from x, cvt f32->bf16) feeds 16 WMMAs whose B
// operands come from ds_load_b128.
// ---------------------------------------------------------------------------
__global__ void __launch_bounds__(256)
gemm_xw_kernel(const float* __restrict__ x,
               const __bf16* __restrict__ bfrag,
               const float* __restrict__ bias,
               float* __restrict__ h,
               int mTiles) {
    __shared__ uint4 Bs[4096];                 // 64 KB: B frags for 4 k-steps

    const int lane  = threadIdx.x & 31;
    const int wib   = threadIdx.x >> 5;        // wave in block: 0..7
    const int hi    = lane >> 4;               // 0: lanes 0-15, 1: lanes 16-31
    const int row16 = lane & 15;

    for (int mtBase = blockIdx.x * 8; mtBase < mTiles; mtBase += gridDim.x * 8) {
        const int  mt     = mtBase + wib;
        const bool active = (mt < mTiles);     // wave-uniform -> EXEC stays all-1s
        v8f acc[NTILES] = {};

        // A-fragment base: row = mt*16 + (lane&15);
        // elems 0-7 <- K kbase+0..7, elems 8-15 <- K kbase+16..23, kbase=8*hi.
        const float* xrow = x + ((long)(active ? mt : 0) * 16 + row16) * DDIM + hi * 8;

        for (int half = 0; half < 2; ++half) {
            __syncthreads();
            // Cooperative stage of 64 KB (contiguous in bfrag): 4096 uint4,
            // 16 per thread, coalesced in global and bank-spread in LDS.
            const uint4* gsrc = (const uint4*)((const char*)bfrag + half * 65536);
#pragma unroll
            for (int i = 0; i < 16; ++i)
                Bs[threadIdx.x + i * 256] = gsrc[threadIdx.x + i * 256];
            __syncthreads();

            if (active) {
#pragma unroll
                for (int ks2 = 0; ks2 < 4; ++ks2) {
                    const float* ap = xrow + (half * 4 + ks2) * 32;
                    v4f a0 = *(const v4f*)(ap + 0);
                    v4f a1 = *(const v4f*)(ap + 4);
                    v4f a2 = *(const v4f*)(ap + 16);
                    v4f a3 = *(const v4f*)(ap + 20);
                    v16bf A;
#pragma unroll
                    for (int j = 0; j < 4; ++j) {
                        A[j]      = f2bf(a0[j]);
                        A[4 + j]  = f2bf(a1[j]);
                        A[8 + j]  = f2bf(a2[j]);
                        A[12 + j] = f2bf(a3[j]);
                    }
                    // B frags in LDS: ((ks2*16 + nt)*32 + lane) 32 B chunks.
                    const v16bf* bp = (const v16bf*)Bs + (ks2 * NTILES * 32 + lane);
#pragma unroll
                    for (int nt = 0; nt < NTILES; ++nt) {
                        acc[nt] = __builtin_amdgcn_wmma_f32_16x16x32_bf16(
                            false, A, false, bp[nt * 32], (short)0, acc[nt],
                            false, false);
                    }
                }
            }
        }

        if (active) {
            // Epilogue: C/D layout -> VGPR i, lane l: row = i + 8*hi, col = l&15.
#pragma unroll
            for (int nt = 0; nt < NTILES; ++nt) {
                float b = bias[nt * 16 + row16];
                float* hp = h + ((long)mt * 16 + hi * 8) * DDIM + nt * 16 + row16;
#pragma unroll
                for (int i = 0; i < 8; ++i) hp[i * DDIM] = acc[nt][i] + b;
            }
        }
    }
}

// ---------------------------------------------------------------------------
// SpMM scatter: one wave per edge. Edge data made wave-uniform
// (readfirstlane -> scalar loads). Lanes gather 1 KB of h[src] contiguously
// (2 x b128/lane, L2-resident) and scatter 8 global_atomic_add_f32 each.
// ---------------------------------------------------------------------------
__global__ void spmm_edge_kernel(const int* __restrict__ esrc,
                                 const int* __restrict__ edst,
                                 const float* __restrict__ eval,
                                 const float* __restrict__ h,
                                 float* __restrict__ out,
                                 int nEdges) {
    const int lane = threadIdx.x & 31;
    const int wid  = __builtin_amdgcn_readfirstlane(
        (int)(blockIdx.x * (blockDim.x >> 5) + (threadIdx.x >> 5)));
    const int nW   = gridDim.x * (blockDim.x >> 5);

    for (int e = wid; e < nEdges; e += nW) {
        int   src = esrc[e];
        int   dst = edst[e];
        float v   = eval[e];

        const v4f* hp = (const v4f*)(h + (long)src * DDIM);
        v4f h0 = hp[lane * 2 + 0];
        v4f h1 = hp[lane * 2 + 1];

        float* op = out + (long)dst * DDIM + lane * 8;
#pragma unroll
        for (int j = 0; j < 4; ++j) unsafeAtomicAdd(op + j,     v * h0[j]);
#pragma unroll
        for (int j = 0; j < 4; ++j) unsafeAtomicAdd(op + 4 + j, v * h1[j]);
    }
}

// ---------------------------------------------------------------------------
extern "C" void kernel_launch(void* const* d_in, const int* in_sizes, int n_in,
                              void* d_out, int out_size, void* d_ws, size_t ws_size,
                              hipStream_t stream) {
    const float* x    = (const float*)d_in[0];
    const float* W    = (const float*)d_in[1];
    const float* bias = (const float*)d_in[2];
    const int*   esrc = (const int*)d_in[3];
    const int*   edst = (const int*)d_in[4];
    const float* eval = (const float*)d_in[5];
    float*       out  = (float*)d_out;

    const int nNodes = in_sizes[0] / DDIM;
    const int nEdges = in_sizes[3];
    const int mTiles = nNodes / 16;            // 100000 / 16 = 6250 (exact)

    // Scratch layout: [0, 128KB) = W bf16 fragments; [128KB, ...) = h.
    __bf16* bfrag = (__bf16*)d_ws;
    float*  h     = (float*)((char*)d_ws + (size_t)(128 << 10));

    // 1) Swizzle W into WMMA B-fragment layout (bf16).
    wfrag_prep_kernel<<<(NFRAG_HALFS + 255) / 256, 256, 0, stream>>>(W, bfrag);

    // 2) Zero output (required every call: atomics accumulate into d_out).
    long n4 = (long)nNodes * (DDIM / 4);
    zero_out_kernel<<<(int)((n4 + 255) / 256), 256, 0, stream>>>((float4*)d_out, n4);

    // 3) Dense transform h = x@W + b via bf16 WMMA (B staged through LDS).
    int gemmBlocks = (mTiles + 7) / 8;         // each block stages B once
    gemm_xw_kernel<<<gemmBlocks, 256, 0, stream>>>(x, bfrag, bias, h, mTiles);

    // 4) COO scatter-add: out[dst] += val * h[src].
    spmm_edge_kernel<<<2048, 256, 0, stream>>>(esrc, edst, eval, h, out, nEdges);
}
